// SimpleUNetPad_17145509446068
// MI455X (gfx1250) — matrix operand order repaired and checked by reading, not compiled, not run on hardware
//
#include <hip/hip_runtime.h>

#define N_IN_   262144
#define N_OUT_  393216
#define M_PTS   262144
#define KOFF    27
#define CIN     32
#define COUT    64
#define BN_EPS  1e-5f

#define TILES_PER_K    (M_PTS / 16)            // 16384 16-point tiles per kernel offset
#define TILES_PER_WAVE 8
#define WAVES_PER_K    (TILES_PER_K / TILES_PER_WAVE)   // 2048

typedef __bf16          v16bf __attribute__((ext_vector_type(16)));
typedef float           v8f   __attribute__((ext_vector_type(8)));
typedef unsigned short  v16u  __attribute__((ext_vector_type(16)));

// fp32 -> bf16 round-to-nearest-even, pure bit math
__device__ __forceinline__ unsigned short f2bf(float f) {
    unsigned int u = __float_as_uint(f);
    u += 0x7FFFu + ((u >> 16) & 1u);
    return (unsigned short)(u >> 16);
}

// ---------------------------------------------------------------------------
// 1) Zero the conv accumulator (d_out) and BN stats (d_ws[0..127])
// ---------------------------------------------------------------------------
__global__ void zero_kernel(float4* __restrict__ acc4, float* __restrict__ stats, long n4) {
    long i = (long)blockIdx.x * blockDim.x + threadIdx.x;
    if (i < n4) acc4[i] = make_float4(0.f, 0.f, 0.f, 0.f);
    if (blockIdx.x == 0 && threadIdx.x < 2 * COUT) stats[threadIdx.x] = 0.f;
}

// ---------------------------------------------------------------------------
// 2) Sparse conv, persistent waves: each wave owns one kernel offset k and
//    loops over 8 M-tiles. B fragments (W[k] as bf16) are loaded ONCE per
//    wave and kept in registers across the tile loop.
//    Control flow is wave-uniform: EXEC stays all-ones for WMMA.
// ---------------------------------------------------------------------------
__global__ __launch_bounds__(256, 1) void spconv_wmma(
        const float* __restrict__ data,      // [N_IN, CIN]
        const float* __restrict__ W,         // [KOFF, CIN, COUT]
        const int*   __restrict__ in_idx,    // [KOFF, M_PTS]
        const int*   __restrict__ out_idx,   // [KOFF, M_PTS]
        float*       __restrict__ acc) {     // [N_OUT, COUT]
    const int lane = threadIdx.x & 31;
    const int wave = threadIdx.x >> 5;
    const int gwave = blockIdx.x * 8 + wave;          // wave-uniform, exact grid
    const int k     = gwave / WAVES_PER_K;
    const int wslot = gwave - k * WAVES_PER_K;

    const int row  = lane & 15;                       // row / column within tile
    const int half = lane >> 4;                       // 0: lanes 0-15, 1: lanes 16-31

    // ---- load B fragments once: W[k] (32x64 fp32) -> 4 x (32x16 bf16) ----
    // B k order used here (contiguous, not AMD's table order): lanes 0-15 hold K=0..15, lanes 16-31 hold
    // K=16..31; column N = lane&15.
    const float* Wk = W + (size_t)k * CIN * COUT;
    v16bf Bf[4];
#pragma unroll
    for (int t = 0; t < 4; ++t) {
        v16u bu;
#pragma unroll
        for (int e = 0; e < 16; ++e)
            bu[e] = f2bf(Wk[(half * 16 + e) * COUT + t * 16 + row]);
        Bf[t] = __builtin_bit_cast(v16bf, bu);
    }

    const int* ii = in_idx  + k * M_PTS;
    const int* oo = out_idx + k * M_PTS;

    for (int it = 0; it < TILES_PER_WAVE; ++it) {     // wave-uniform loop
        const int mbase = (wslot * TILES_PER_WAVE + it) * 16;

        // ---- gather one data row per (lane&15), build A (16x32 bf16) ----
        // A k order used here: lanes 0-15 hold K=0..15, lanes 16-31 hold
        // K=16..31 (contiguous, matching B).
        const int g = ii[mbase + row];
        const float4* src4 = reinterpret_cast<const float4*>(data + (size_t)g * CIN);
        float4 c0 = src4[half * 4 + 0];               // K = half*16 + 0..3
        float4 c1 = src4[half * 4 + 1];               // K = half*16 + 4..7
        float4 c2 = src4[half * 4 + 2];               // K = half*16 + 8..11
        float4 c3 = src4[half * 4 + 3];               // K = half*16 + 12..15
        v16u au;
        au[0]  = f2bf(c0.x); au[1]  = f2bf(c0.y); au[2]  = f2bf(c0.z); au[3]  = f2bf(c0.w);
        au[4]  = f2bf(c1.x); au[5]  = f2bf(c1.y); au[6]  = f2bf(c1.z); au[7]  = f2bf(c1.w);
        au[8]  = f2bf(c2.x); au[9]  = f2bf(c2.y); au[10] = f2bf(c2.z); au[11] = f2bf(c2.w);
        au[12] = f2bf(c3.x); au[13] = f2bf(c3.y); au[14] = f2bf(c3.z); au[15] = f2bf(c3.w);
        v16bf A = __builtin_bit_cast(v16bf, au);

        // ---- output row indices for this lane's C/D rows (2 x b128) ----
        // C/D f32 16x16: VGPR r -> M=r (lanes 0-15) or M=r+8 (lanes 16-31)
        const int4* op = reinterpret_cast<const int4*>(oo + mbase + half * 8);
        int4 o0 = op[0];
        int4 o1 = op[1];
        int oi[8] = { o0.x, o0.y, o0.z, o0.w, o1.x, o1.y, o1.z, o1.w };

#pragma unroll
        for (int t = 0; t < 4; ++t) {                 // 4 N-tiles cover COUT=64
            v8f C = {};
            C = __builtin_amdgcn_wmma_f32_16x16x32_bf16(
                    /*neg_a=*/false, A, /*neg_b=*/false, Bf[t],
                    /*c_mod=*/(short)0, C, /*reuse_a=*/false, /*reuse_b=*/false);

            const int col = t * 16 + row;
#pragma unroll
            for (int r = 0; r < 8; ++r)
                __hip_atomic_fetch_add(&acc[(size_t)oi[r] * COUT + col], C[r],
                                       __ATOMIC_RELAXED, __HIP_MEMORY_SCOPE_AGENT);
        }
    }
}

// ---------------------------------------------------------------------------
// 3) Per-channel sum / sum-of-squares (LDS reduce, 1 atomic per channel/block)
// ---------------------------------------------------------------------------
__global__ __launch_bounds__(256) void stats_kernel(const float* __restrict__ acc,
                                                    float* __restrict__ stats) {
    const int ch = threadIdx.x & (COUT - 1);
    const int rg = threadIdx.x >> 6;                  // 0..3 row groups per block
    float s = 0.f, q = 0.f;
    for (int r = blockIdx.x * 4 + rg; r < N_OUT_; r += gridDim.x * 4) {
        float v = acc[(size_t)r * COUT + ch];
        s += v;
        q += v * v;
    }
    __shared__ float ls[256], lq[256];
    ls[threadIdx.x] = s;
    lq[threadIdx.x] = q;
    __syncthreads();
    if (threadIdx.x < COUT) {
        float ts = ls[threadIdx.x] + ls[threadIdx.x + 64] + ls[threadIdx.x + 128] + ls[threadIdx.x + 192];
        float tq = lq[threadIdx.x] + lq[threadIdx.x + 64] + lq[threadIdx.x + 128] + lq[threadIdx.x + 192];
        __hip_atomic_fetch_add(&stats[threadIdx.x],        ts, __ATOMIC_RELAXED, __HIP_MEMORY_SCOPE_AGENT);
        __hip_atomic_fetch_add(&stats[COUT + threadIdx.x], tq, __ATOMIC_RELAXED, __HIP_MEMORY_SCOPE_AGENT);
    }
}

// ---------------------------------------------------------------------------
// 4) In-place batch norm
// ---------------------------------------------------------------------------
__global__ __launch_bounds__(256) void bn_kernel(float* __restrict__ out,
                                                 const float* __restrict__ stats,
                                                 const float* __restrict__ gamma,
                                                 const float* __restrict__ beta) {
    size_t i = (size_t)blockIdx.x * blockDim.x + threadIdx.x;
    if (i >= (size_t)N_OUT_ * COUT) return;
    const int ch = (int)(i & (COUT - 1));
    const float invN = 1.0f / (float)N_OUT_;
    const float mean = stats[ch] * invN;
    const float var  = stats[COUT + ch] * invN - mean * mean;
    const float inv  = rsqrtf(var + BN_EPS);
    out[i] = (out[i] - mean) * inv * gamma[ch] + beta[ch];
}

// ---------------------------------------------------------------------------
extern "C" void kernel_launch(void* const* d_in, const int* in_sizes, int n_in,
                              void* d_out, int out_size, void* d_ws, size_t ws_size,
                              hipStream_t stream) {
    const float* data    = (const float*)d_in[0];
    const float* W       = (const float*)d_in[1];
    const float* gamma   = (const float*)d_in[2];
    const float* beta    = (const float*)d_in[3];
    const int*   in_idx  = (const int*)d_in[4];
    const int*   out_idx = (const int*)d_in[5];
    float* out   = (float*)d_out;                     // fp32 accumulator, BN'd in place
    float* stats = (float*)d_ws;                      // 128 floats: [sum(64), sumsq(64)]

    const long n4 = (long)N_OUT_ * COUT / 4;
    zero_kernel<<<(int)((n4 + 255) / 256), 256, 0, stream>>>((float4*)out, stats, n4);

    const int nWaves = KOFF * WAVES_PER_K;            // 55296, divisible by 8
    spconv_wmma<<<nWaves / 8, 256, 0, stream>>>(data, W, in_idx, out_idx, out);

    stats_kernel<<<2048, 256, 0, stream>>>(out, stats);

    const long nElem = (long)N_OUT_ * COUT;
    bn_kernel<<<(int)((nElem + 255) / 256), 256, 0, stream>>>(out, stats, gamma, beta);
}
